// HybridMemory_63745904607634
// MI455X (gfx1250) — matrix operand order repaired and checked by reading, not compiled, not run on hardware
//
#include <hip/hip_runtime.h>
#include <cstdint>
#include <cstddef>

// Problem constants (match reference)
#define DIMK     2048
#define NCOLS    65536
#define MROWS    256
#define KSTEPS   (DIMK / 32)      // 64 k-steps of 32
#define MT_WAVE  8                // m-tiles (of 16 rows) accumulated per wave
#define NT_BLOCK 8                // one 16-wide n-tile per wave, 8 waves/block

typedef __attribute__((ext_vector_type(16))) __bf16 v16bf;
typedef __attribute__((ext_vector_type(8)))  float  v8f;

union Frag32 {                    // 16 bf16 halves = 8 dwords = one WMMA operand
    uint32_t u[8];
    uint4    q[2];
    v16bf    bf;
};

// Split 16 fp32 into bf16 hi (truncate) and bf16 lo (residual), packed 2/dword.
__device__ __forceinline__ void split_bf16x2(const float* f, uint32_t* hi, uint32_t* lo) {
#pragma unroll
    for (int p = 0; p < 8; ++p) {
        float    f0 = f[2 * p + 0], f1 = f[2 * p + 1];
        uint32_t u0 = __float_as_uint(f0);
        uint32_t u1 = __float_as_uint(f1);
        uint32_t h0 = u0 & 0xFFFF0000u;
        uint32_t h1 = u1 & 0xFFFF0000u;
        hi[p] = (u0 >> 16) | h1;
        float r0 = f0 - __uint_as_float(h0);   // exact residual
        float r1 = f1 - __uint_as_float(h1);
        lo[p] = (__float_as_uint(r0) >> 16) | (__float_as_uint(r1) & 0xFFFF0000u);
    }
}

__device__ __forceinline__ void unpack16(float4 a, float4 b, float4 c, float4 d, float* f) {
    f[0] = a.x;  f[1] = a.y;  f[2]  = a.z;  f[3]  = a.w;
    f[4] = b.x;  f[5] = b.y;  f[6]  = b.z;  f[7]  = b.w;
    f[8] = c.x;  f[9] = c.y;  f[10] = c.z;  f[11] = c.w;
    f[12] = d.x; f[13] = d.y; f[14] = d.z;  f[15] = d.w;
}

#define WMMA_BF16(A, B, C) \
    __builtin_amdgcn_wmma_f32_16x16x32_bf16(false, (A), false, (B), (short)0, (C), false, false)

// ---------------------------------------------------------------------------
// Pre-kernel: convert inputs (256x2048 fp32) into WMMA-A-fragment-ordered
// bf16 hi/lo arrays. One thread per (mtile, kstep, lane): writes the 16 halves
// that lane feeds to v_wmma_f32_16x16x32_bf16.
//   A layout (16-bit, 16x32): lane<16 -> K = kb..kb+7, kb+16..kb+23,
//   kb = s*32 + (lane>>4)*8.  Fragment uint4 index = ((t*64+s)*32 + lane)*2.
// ---------------------------------------------------------------------------
__global__ __launch_bounds__(256) void hm_preconv_a(const float* __restrict__ inputs,
                                                    uint32_t* __restrict__ ahi,
                                                    uint32_t* __restrict__ alo) {
    int tid  = blockIdx.x * 256 + threadIdx.x;      // == (t*64 + s)*32 + lane
    int lane = tid & 31;
    int kb   = ((tid >> 5) & 63) * 32 + (lane >> 4) * 8;
    int m    = (tid >> 11) * 16 + (lane & 15);

    const float*  src = inputs + (size_t)m * DIMK + kb;
    const float4* s0  = (const float4*)src;         // K = kb .. kb+7
    const float4* s1  = (const float4*)(src + 16);  // K = kb+16 .. kb+23
    float f[16];
    unpack16(s0[0], s0[1], s1[0], s1[1], f);

    Frag32 hi, lo;
    split_bf16x2(f, hi.u, lo.u);

    uint4* dh = (uint4*)ahi + (size_t)tid * 2;
    uint4* dl = (uint4*)alo + (size_t)tid * 2;
    dh[0] = hi.q[0]; dh[1] = hi.q[1];
    dl[0] = lo.q[0]; dl[1] = lo.q[1];
}

// ---------------------------------------------------------------------------
// Main GEMM: out[m,n] = sum_k inputs[m,k]*features[n,k] via bf16x3 WMMA.
// Grid (mblock=2, nchunk=512), block 256 = 8 waves. Wave w: n-tile nc*8+w,
// accumulates 8 m-tiles (128 rows). A hi/lo fragments staged through LDS
// (double-buffered, cooperative: all 8 waves share identical A data).
// ---------------------------------------------------------------------------
__global__ __launch_bounds__(256) void hm_gemm_lds(const float* __restrict__ features,
                                                   const uint32_t* __restrict__ ahi,
                                                   const uint32_t* __restrict__ alo,
                                                   float* __restrict__ out) {
    // [buf][plane][tile][lane]: plane 0/1 = ahi q0/q1, plane 2/3 = alo q0/q1.
    // For a (buf,plane,tile) read: lane-contiguous 16B stride -> conflict-free.
    __shared__ uint4 lds[2][4][MT_WAVE][32];        // 32 KB

    const int lane = threadIdx.x & 31;
    const int wave = threadIdx.x >> 5;
    const int row  = lane & 15;
    const int half = lane >> 4;
    const int mblock = blockIdx.x;                  // 0..1
    const int n0     = (blockIdx.y * NT_BLOCK + wave) * 16;

    // Staging role of this thread: tile (= wave), lane. Per-s stride = 64 uint4.
    const int    gt_st = mblock * MT_WAVE + wave;
    const uint4* gh    = (const uint4*)ahi + ((size_t)(gt_st * KSTEPS) * 32 + lane) * 2;
    const uint4* gl    = (const uint4*)alo + ((size_t)(gt_st * KSTEPS) * 32 + lane) * 2;

    // B operand: lane holds column n0+row, K = s*32 + half*8 + (0..7, 16..23).
    const float* bp = features + (size_t)(n0 + row) * DIMK + half * 8;

    v8f acc[MT_WAVE];
    const v8f vzero = {};
#pragma unroll
    for (int t = 0; t < MT_WAVE; ++t) acc[t] = vzero;

    // ---- prologue: stage s=0 into buffer 0, preload B(s=0) ----
    {
        uint4 h0 = gh[0], h1 = gh[1], l0 = gl[0], l1 = gl[1];
        lds[0][0][wave][lane] = h0;
        lds[0][1][wave][lane] = h1;
        lds[0][2][wave][lane] = l0;
        lds[0][3][wave][lane] = l1;
    }
    const float4* b4 = (const float4*)bp;
    float4 q0 = b4[0], q1 = b4[1], q2 = b4[4], q3 = b4[5];
    __syncthreads();

#pragma unroll 1
    for (int s = 0; s < KSTEPS; ++s) {
        const int buf = s & 1;

        // ---- issue next-step staging + B loads first (long overlap) ----
        uint4  nh0, nh1, nl0, nl1;
        float4 p0, p1, p2, p3;
        if (s + 1 < KSTEPS) {
            const uint4* gh1 = gh + (size_t)(s + 1) * 64;
            const uint4* gl1 = gl + (size_t)(s + 1) * 64;
            nh0 = gh1[0]; nh1 = gh1[1];
            nl0 = gl1[0]; nl1 = gl1[1];
            const float4* bn = (const float4*)(bp + (s + 1) * 32);
            p0 = bn[0]; p1 = bn[1]; p2 = bn[4]; p3 = bn[5];
        }

        // ---- convert current B tile (reused by 8 m-tiles / 24 WMMAs) ----
        float fb[16];
        unpack16(q0, q1, q2, q3, fb);
        Frag32 bhi, blo;
        split_bf16x2(fb, bhi.u, blo.u);

        // ---- compute: A fragments from LDS (low latency) ----
#pragma unroll
        for (int t = 0; t < MT_WAVE; ++t) {
            Frag32 fahi, falo;
            fahi.q[0] = lds[buf][0][t][lane];
            fahi.q[1] = lds[buf][1][t][lane];
            falo.q[0] = lds[buf][2][t][lane];
            falo.q[1] = lds[buf][3][t][lane];
            // bf16x3: hi*hi + hi*lo + lo*hi (lo*lo ~2^-32, dropped)
            acc[t] = WMMA_BF16(fahi.bf, bhi.bf, acc[t]);
            acc[t] = WMMA_BF16(fahi.bf, blo.bf, acc[t]);
            acc[t] = WMMA_BF16(falo.bf, bhi.bf, acc[t]);
        }

        // ---- write next-step stage into the other buffer ----
        if (s + 1 < KSTEPS) {
            lds[buf ^ 1][0][wave][lane] = nh0;
            lds[buf ^ 1][1][wave][lane] = nh1;
            lds[buf ^ 1][2][wave][lane] = nl0;
            lds[buf ^ 1][3][wave][lane] = nl1;
            q0 = p0; q1 = p1; q2 = p2; q3 = p3;
        }
        __syncthreads();   // protects buffer reuse both directions
    }

    // ---- store: C/D layout: VGPR r, lanes<16 -> M=r, lanes>=16 -> M=8+r ----
    const int mbase = mblock * (MT_WAVE * 16) + half * 8;
    float* op = out + (size_t)(n0 + row);
#pragma unroll
    for (int t = 0; t < MT_WAVE; ++t) {
#pragma unroll
        for (int r = 0; r < 8; ++r) {
            op[(size_t)(mbase + t * 16 + r) * NCOLS] = acc[t][r];
        }
    }
}

// ---------------------------------------------------------------------------
// Fallback (workspace too small): inline A conversion, register-only.
// ---------------------------------------------------------------------------
__global__ __launch_bounds__(256) void hm_gemm_nows(const float* __restrict__ features,
                                                    const float* __restrict__ inputs,
                                                    float* __restrict__ out) {
    const int lane = threadIdx.x & 31;
    const int wave = threadIdx.x >> 5;
    const int row  = lane & 15;
    const int half = lane >> 4;
    const int mblock = blockIdx.x;
    const int n0     = (blockIdx.y * NT_BLOCK + wave) * 16;

    const float* bp = features + (size_t)(n0 + row) * DIMK + half * 8;

    v8f acc[MT_WAVE];
    const v8f vzero = {};
#pragma unroll
    for (int t = 0; t < MT_WAVE; ++t) acc[t] = vzero;

#pragma unroll 1
    for (int s = 0; s < KSTEPS; ++s) {
        const float4* b4 = (const float4*)(bp + s * 32);
        float fb[16];
        unpack16(b4[0], b4[1], b4[4], b4[5], fb);
        Frag32 bhi, blo;
        split_bf16x2(fb, bhi.u, blo.u);

#pragma unroll
        for (int t = 0; t < MT_WAVE; ++t) {
            const int gt = mblock * MT_WAVE + t;
            const float*  apf = inputs + (size_t)(gt * 16 + row) * DIMK + s * 32 + half * 8;
            const float4* a4  = (const float4*)apf;
            const float4* a4b = (const float4*)(apf + 16);
            float fa[16];
            unpack16(a4[0], a4[1], a4b[0], a4b[1], fa);
            Frag32 fahi, falo;
            split_bf16x2(fa, fahi.u, falo.u);
            acc[t] = WMMA_BF16(fahi.bf, bhi.bf, acc[t]);
            acc[t] = WMMA_BF16(fahi.bf, blo.bf, acc[t]);
            acc[t] = WMMA_BF16(falo.bf, bhi.bf, acc[t]);
        }
    }

    const int mbase = mblock * (MT_WAVE * 16) + half * 8;
    float* op = out + (size_t)(n0 + row);
#pragma unroll
    for (int t = 0; t < MT_WAVE; ++t) {
#pragma unroll
        for (int r = 0; r < 8; ++r) {
            op[(size_t)(mbase + t * 16 + r) * NCOLS] = acc[t][r];
        }
    }
}

// ---------------------------------------------------------------------------
extern "C" void kernel_launch(void* const* d_in, const int* in_sizes, int n_in,
                              void* d_out, int out_size, void* d_ws, size_t ws_size,
                              hipStream_t stream) {
    const float* inputs   = (const float*)d_in[0];   // [256, 2048] fp32
    const float* features = (const float*)d_in[1];   // [65536, 2048] fp32
    float*       out      = (float*)d_out;           // [256, 65536] fp32
    (void)in_sizes; (void)n_in; (void)out_size;

    // A fragments: 16 mtiles * 64 ksteps * 32 lanes * 16 halves * 2B = 1 MB each
    const size_t frag_dwords = (size_t)(MROWS / 16) * KSTEPS * 32 * 8;   // 262144
    const size_t ws_needed   = frag_dwords * 4 * 2;                      // 2 MB

    dim3 grid(2, NCOLS / (16 * NT_BLOCK), 1);   // (mblock, nchunk) = (2, 512)
    dim3 block(256, 1, 1);

    if (ws_size >= ws_needed) {
        uint32_t* ahi = (uint32_t*)d_ws;
        uint32_t* alo = ahi + frag_dwords;
        hm_preconv_a<<<(MROWS / 16) * KSTEPS * 32 / 256, 256, 0, stream>>>(inputs, ahi, alo);
        hm_gemm_lds<<<grid, block, 0, stream>>>(features, ahi, alo, out);
    } else {
        hm_gemm_nows<<<grid, block, 0, stream>>>(features, inputs, out);
    }
}
